// MoERagged_68796786147589
// MI455X (gfx1250) — matrix operand order repaired and checked by reading, not compiled, not run on hardware
//
#include <hip/hip_runtime.h>
#include <math.h>
#include <stdint.h>

typedef __bf16 bf16_t;
typedef __attribute__((ext_vector_type(16))) __bf16 v16bf;
typedef __attribute__((ext_vector_type(8)))  __bf16 v8bf;
typedef __attribute__((ext_vector_type(8)))  float  v8f;

#define G_DIM 4
#define S_DIM 2048
#define F_DIM 2048
#define E_DIM 8
#define H_DIM 2048
#define NTOK  (G_DIM * S_DIM)              // 8192 tokens
#define NROWS (NTOK * 2)                   // 16384 token-expert rows
#define CAP   (NROWS + E_DIM * 128)        // padded capacity: 17408
#define MT    (CAP / 128)                  // 136 M-tiles
#define LDT   40                           // LDS row stride (bf16), keeps 16B align

// ---------------------------------------------------------------- helpers
// branch-free tanh-approx GELU: tanh(z) = 1 - 2/(exp(2z)+1) via v_exp_f32
__device__ __forceinline__ float gelu_tanh(float x) {
    const float c = 0.7978845608028654f; // sqrt(2/pi)
    const float z = c * (x + 0.044715f * x * x * x);
    const float t = 1.0f - 2.0f / (__expf(2.0f * z) + 1.0f);
    return 0.5f * x * (1.0f + t);
}

// ---------------------------------------------------------------- router
__global__ __launch_bounds__(256) void router_kernel(
    const float* __restrict__ x, const float* __restrict__ rlogits,
    const float* __restrict__ rscale,
    int* __restrict__ topk_idx, float* __restrict__ topk_w,
    int* __restrict__ counts)
{
    const int t   = blockIdx.x;
    const int tid = threadIdx.x;
    const float* xr = x + (size_t)t * F_DIM;

    __shared__ float red[256];
    __shared__ float lg[E_DIM];

    float xv[8];
    float ss = 0.f;
#pragma unroll
    for (int i = 0; i < 8; ++i) {
        xv[i] = xr[tid + 256 * i];
        ss += xv[i] * xv[i];
    }
    red[tid] = ss;
    __syncthreads();
    for (int off = 128; off > 0; off >>= 1) {
        if (tid < off) red[tid] += red[tid + off];
        __syncthreads();
    }
    const float inv = rsqrtf(red[0] * (1.0f / F_DIM) + 1e-6f) * rsqrtf((float)F_DIM);
    __syncthreads();

    float p[E_DIM];
#pragma unroll
    for (int e = 0; e < E_DIM; ++e) p[e] = 0.f;
#pragma unroll
    for (int i = 0; i < 8; ++i) {
        const int f = tid + 256 * i;
        const float xin = xv[i] * inv * rscale[f];
        const float* r = rlogits + (size_t)f * E_DIM;
#pragma unroll
        for (int e = 0; e < E_DIM; ++e) p[e] += xin * r[e];
    }
#pragma unroll
    for (int e = 0; e < E_DIM; ++e) {
        red[tid] = p[e];
        __syncthreads();
        for (int off = 128; off > 0; off >>= 1) {
            if (tid < off) red[tid] += red[tid + off];
            __syncthreads();
        }
        if (tid == 0) lg[e] = red[0];
        __syncthreads();
    }

    if (tid == 0) {
        float mx = lg[0];
#pragma unroll
        for (int e = 1; e < E_DIM; ++e) mx = fmaxf(mx, lg[e]);
        float ex[E_DIM], sum = 0.f;
#pragma unroll
        for (int e = 0; e < E_DIM; ++e) { ex[e] = __expf(lg[e] - mx); sum += ex[e]; }
        int i0 = 0; float b0 = lg[0];
#pragma unroll
        for (int e = 1; e < E_DIM; ++e) if (lg[e] > b0) { b0 = lg[e]; i0 = e; }
        int i1 = -1; float b1 = -3.4e38f;
#pragma unroll
        for (int e = 0; e < E_DIM; ++e) if (e != i0 && lg[e] > b1) { b1 = lg[e]; i1 = e; }
        const float p0 = ex[i0] / sum, p1 = ex[i1] / sum;
        const float rn = p0 + p1;
        topk_idx[2 * t + 0] = i0;
        topk_idx[2 * t + 1] = i1;
        topk_w[2 * t + 0]   = p0 / rn;
        topk_w[2 * t + 1]   = p1 / rn;
        atomicAdd(&counts[i0], 1);
        atomicAdd(&counts[i1], 1);
    }
}

// ------------------------------------------------------- row bookkeeping
__global__ __launch_bounds__(256) void init_rows_kernel(int* __restrict__ row_token)
{
    const int i = blockIdx.x * blockDim.x + threadIdx.x;
    if (i < CAP) row_token[i] = -1;
}

__global__ void scan_kernel(const int* __restrict__ counts,
                            int* __restrict__ base, int* __restrict__ cursor)
{
    if (threadIdx.x == 0) {
        int acc = 0;
        for (int e = 0; e < E_DIM; ++e) {
            base[e]   = acc;
            cursor[e] = acc;
            acc += (counts[e] + 127) & ~127;   // pad each expert segment to 128 rows
        }
        base[E_DIM] = acc;                     // padded_total
    }
}

__global__ __launch_bounds__(256) void assign_kernel(
    const int* __restrict__ topk_idx, const float* __restrict__ topk_w,
    const float* __restrict__ pes, int* __restrict__ cursor,
    int* __restrict__ row_token, float* __restrict__ row_scale)
{
    const int t = blockIdx.x * blockDim.x + threadIdx.x;
    if (t >= NTOK) return;
#pragma unroll
    for (int kk = 0; kk < 2; ++kk) {
        const int   e = topk_idx[2 * t + kk];
        const float w = topk_w[2 * t + kk];
        const int pos = atomicAdd(&cursor[e], 1);
        row_token[pos] = t;
        row_scale[pos] = w * pes[e];
    }
}

// ---------------------------------------------------------- WMMA helpers
__device__ __forceinline__ v16bf frag_a(const bf16_t* row, int lane) {
    // 16-bit A 16x32: lanes 0-15 hold K {0..7,16..23}; lanes 16-31 hold K {8..15,24..31}
    const int ks = (lane >> 4) * 8;
    v8bf lo = *(const v8bf*)(row + ks);
    v8bf hi = *(const v8bf*)(row + ks + 16);
    return __builtin_shufflevector(lo, hi, 0,1,2,3,4,5,6,7,8,9,10,11,12,13,14,15);
}
__device__ __forceinline__ v16bf frag_b(const bf16_t* row, int lane) {
    // 16-bit B 32x16: lanes 0-15 hold K {0..7,16..23}; lanes 16-31 hold K {8..15,24..31}
    const int ks = (lane >> 4) * 8;
    v8bf lo = *(const v8bf*)(row + ks);
    v8bf hi = *(const v8bf*)(row + ks + 16);
    return __builtin_shufflevector(lo, hi, 0,1,2,3,4,5,6,7,8,9,10,11,12,13,14,15);
}
__device__ __forceinline__ v8f wmma_bf16(v16bf a, v16bf b, v8f c) {
    return __builtin_amdgcn_wmma_f32_16x16x32_bf16(false, a, false, b, (short)0, c,
                                                   false, false);
}
// async copy 16B global -> LDS (CDNA5, ASYNCcnt-tracked)
__device__ __forceinline__ void async_b128(bf16_t* lds, const bf16_t* g) {
    const unsigned lds_off = (unsigned)(uintptr_t)lds;
    asm volatile("global_load_async_to_lds_b128 %0, %1, off"
                 :: "v"(lds_off), "v"(g) : "memory");
}
__device__ __forceinline__ void wait_async() {
    asm volatile("s_wait_asynccnt 0" ::: "memory");
}

// -------------------------------------------- GEMM1: gate/up + GELU fuse
// Block tile: 128 rows x (64 gate + 64 up cols), BK=32, double-buffered LDS.
// 8 waves in 4x2: wave owns 32 rows x (32 gate + 32 up) -> 2 A frags x 4 B frags.
__global__ __launch_bounds__(256) void gemm_gate_kernel(
    const float* __restrict__ x, const float* __restrict__ wgate,
    const int* __restrict__ row_token, const int* __restrict__ base,
    bf16_t* __restrict__ act)
{
    __shared__ bf16_t Alds[2][128 * LDT];
    __shared__ bf16_t Blds[2][128 * LDT];

    const int m0 = blockIdx.x * 128;
    const int padded_total = base[E_DIM];
    if (m0 >= padded_total) return;
    int e = 0;
    while (e < E_DIM - 1 && m0 >= base[e + 1]) ++e;

    const int n0   = blockIdx.y * 64;   // gate column base (up cols at +H)
    const int tid  = threadIdx.x;
    const int lane = tid & 31;
    const int w    = tid >> 5;
    const int rg   = w >> 1;            // row group 0..3
    const int cg   = w & 1;             // col group 0..1
    const int lr   = lane & 15;

    const int r0 = tid >> 3;            // 0..31
    const int c4 = (tid & 7) * 4;       // 0..28
    int tokr[4];
#pragma unroll
    for (int it = 0; it < 4; ++it) tokr[it] = row_token[m0 + r0 + 32 * it];

    const size_t gbase = (size_t)e * 2 * H_DIM * F_DIM;
    float4 aReg[4], bReg[4];

    auto brow_ptr = [&](int it, int k0) -> const float* {
        const int r = r0 + 32 * it;
        const size_t row = (r < 64) ? (size_t)(n0 + r)
                                    : (size_t)(H_DIM + n0 + r - 64);
        return wgate + gbase + row * F_DIM + k0 + c4;
    };
    auto issue_loads = [&](int k0) {
#pragma unroll
        for (int it = 0; it < 4; ++it) {
            float4 av = make_float4(0.f, 0.f, 0.f, 0.f);
            if (tokr[it] >= 0)
                av = *(const float4*)(x + (size_t)tokr[it] * F_DIM + k0 + c4);
            aReg[it] = av;
            bReg[it] = *(const float4*)brow_ptr(it, k0);
        }
    };
    auto store_tiles = [&](int buf) {
#pragma unroll
        for (int it = 0; it < 4; ++it) {
            const int r = r0 + 32 * it;
            bf16_t* da = &Alds[buf][r * LDT + c4];
            da[0] = (bf16_t)aReg[it].x; da[1] = (bf16_t)aReg[it].y;
            da[2] = (bf16_t)aReg[it].z; da[3] = (bf16_t)aReg[it].w;
            bf16_t* db = &Blds[buf][r * LDT + c4];
            db[0] = (bf16_t)bReg[it].x; db[1] = (bf16_t)bReg[it].y;
            db[2] = (bf16_t)bReg[it].z; db[3] = (bf16_t)bReg[it].w;
        }
    };

    v8f acc[2][4];
#pragma unroll
    for (int mt = 0; mt < 2; ++mt)
#pragma unroll
        for (int bt = 0; bt < 4; ++bt)
            acc[mt][bt] = (v8f){0.f,0.f,0.f,0.f,0.f,0.f,0.f,0.f};

    issue_loads(0);
    store_tiles(0);
    __syncthreads();

    int buf = 0;
    for (int k0 = 0; k0 < F_DIM; k0 += 32, buf ^= 1) {
        const bool more = (k0 + 32) < F_DIM;
        if (more) issue_loads(k0 + 32);          // overlap with compute below
        if (k0 + 64 < F_DIM) {                   // keep weight stream warm
#pragma unroll
            for (int it = 0; it < 4; ++it)
                __builtin_prefetch(brow_ptr(it, k0 + 64), 0, 3);
        }

        v16bf af[2], bfv[4];
#pragma unroll
        for (int mt = 0; mt < 2; ++mt)
            af[mt] = frag_a(&Alds[buf][(32 * rg + 16 * mt + lr) * LDT], lane);
#pragma unroll
        for (int bt = 0; bt < 2; ++bt) {
            bfv[bt]     = frag_b(&Blds[buf][(32 * cg + 16 * bt + lr) * LDT], lane);
            bfv[bt + 2] = frag_b(&Blds[buf][(64 + 32 * cg + 16 * bt + lr) * LDT], lane);
        }
#pragma unroll
        for (int mt = 0; mt < 2; ++mt)
#pragma unroll
            for (int bt = 0; bt < 4; ++bt)
                acc[mt][bt] = wmma_bf16(af[mt], bfv[bt], acc[mt][bt]);

        if (more) store_tiles(buf ^ 1);
        __syncthreads();
    }

    // epilogue: act = gelu(gate) * up ; B tile bt pairs with bt+2
    const int mrow = m0 + 32 * rg + 8 * (lane >> 4);
    const int ncol = n0 + 32 * cg + lr;
#pragma unroll
    for (int mt = 0; mt < 2; ++mt) {
#pragma unroll
        for (int bt = 0; bt < 2; ++bt) {
#pragma unroll
            for (int j = 0; j < 8; ++j) {
                const float a = gelu_tanh(acc[mt][bt][j]) * acc[mt][bt + 2][j];
                act[(size_t)(mrow + 16 * mt + j) * H_DIM + (ncol + 16 * bt)] = (bf16_t)a;
            }
        }
    }
}

// ------------------------------------------------ GEMM2: down proj + combine
// Block tile: 128 x 128, BK=32, double-buffered LDS. A tile is a raw bf16 copy
// done with CDNA5 async global->LDS loads; B tile is fp32->bf16 + transpose.
__global__ __launch_bounds__(256) void gemm_out_kernel(
    const bf16_t* __restrict__ act, const float* __restrict__ lin,
    const int* __restrict__ row_token, const float* __restrict__ row_scale,
    const int* __restrict__ base, float* __restrict__ out)
{
    __shared__ bf16_t Alds[2][128 * LDT];
    __shared__ bf16_t Blds[2][128 * LDT];

    const int m0 = blockIdx.x * 128;
    const int padded_total = base[E_DIM];
    if (m0 >= padded_total) return;
    int e = 0;
    while (e < E_DIM - 1 && m0 >= base[e + 1]) ++e;

    const int nb0  = blockIdx.y * 128;
    const int tid  = threadIdx.x;
    const int lane = tid & 31;
    const int w    = tid >> 5;
    const int rg   = w >> 1;
    const int cg   = w & 1;
    const int lr   = lane & 15;

    const int ar = tid >> 2;            // 0..63
    const int ac = (tid & 3) * 8;       // 0,8,16,24
    const int bc4 = (tid & 31) * 4;     // B col 0..124
    const int bk0 = tid >> 5;           // B k row 0..7

    const size_t lbase = (size_t)e * H_DIM * F_DIM;
    float4 bReg[4];

    auto async_copy_A = [&](int buf, int k0) {
#pragma unroll
        for (int it = 0; it < 2; ++it) {
            const int r = ar + 64 * it;
            async_b128(&Alds[buf][r * LDT + ac],
                       act + (size_t)(m0 + r) * H_DIM + k0 + ac);
        }
    };
    auto issue_b = [&](int k0) {
#pragma unroll
        for (int it = 0; it < 4; ++it) {
            const int kk = bk0 + 8 * it;
            bReg[it] = *(const float4*)(lin + lbase + (size_t)(k0 + kk) * F_DIM + nb0 + bc4);
        }
    };
    auto store_b = [&](int buf) {
#pragma unroll
        for (int it = 0; it < 4; ++it) {
            const int kk = bk0 + 8 * it;
            Blds[buf][(bc4 + 0) * LDT + kk] = (bf16_t)bReg[it].x;
            Blds[buf][(bc4 + 1) * LDT + kk] = (bf16_t)bReg[it].y;
            Blds[buf][(bc4 + 2) * LDT + kk] = (bf16_t)bReg[it].z;
            Blds[buf][(bc4 + 3) * LDT + kk] = (bf16_t)bReg[it].w;
        }
    };

    v8f acc[2][4];
#pragma unroll
    for (int mt = 0; mt < 2; ++mt)
#pragma unroll
        for (int bt = 0; bt < 4; ++bt)
            acc[mt][bt] = (v8f){0.f,0.f,0.f,0.f,0.f,0.f,0.f,0.f};

    async_copy_A(0, 0);
    issue_b(0);
    store_b(0);
    wait_async();
    __syncthreads();

    int buf = 0;
    for (int k0 = 0; k0 < H_DIM; k0 += 32, buf ^= 1) {
        const bool more = (k0 + 32) < H_DIM;
        if (more) {
            async_copy_A(buf ^ 1, k0 + 32);      // fully async next A tile
            issue_b(k0 + 32);                    // stage next B in regs
        }
        if (k0 + 64 < H_DIM) {                   // keep weight stream warm
#pragma unroll
            for (int it = 0; it < 4; ++it)
                __builtin_prefetch(
                    lin + lbase + (size_t)(k0 + 64 + bk0 + 8 * it) * F_DIM + nb0 + bc4,
                    0, 3);
        }

        v16bf af[2], bfv[4];
#pragma unroll
        for (int mt = 0; mt < 2; ++mt)
            af[mt] = frag_a(&Alds[buf][(32 * rg + 16 * mt + lr) * LDT], lane);
#pragma unroll
        for (int bt = 0; bt < 4; ++bt)
            bfv[bt] = frag_b(&Blds[buf][(64 * cg + 16 * bt + lr) * LDT], lane);
#pragma unroll
        for (int mt = 0; mt < 2; ++mt)
#pragma unroll
            for (int bt = 0; bt < 4; ++bt)
                acc[mt][bt] = wmma_bf16(af[mt], bfv[bt], acc[mt][bt]);

        if (more) store_b(buf ^ 1);
        wait_async();
        __syncthreads();
    }

    // epilogue: scale + atomic combine into out (2 commutative adds per element)
    const int mrow = m0 + 32 * rg + 8 * (lane >> 4);
    const int ncol = nb0 + 64 * cg + lr;
#pragma unroll
    for (int mt = 0; mt < 2; ++mt) {
#pragma unroll
        for (int j = 0; j < 8; ++j) {
            const int row = mrow + 16 * mt + j;
            const int tok = row_token[row];
            if (tok < 0) continue;
            const float scl = row_scale[row];
            float* orow = out + (size_t)tok * F_DIM;
#pragma unroll
            for (int bt = 0; bt < 4; ++bt) {
                unsafeAtomicAdd(orow + ncol + 16 * bt, acc[mt][bt][j] * scl);
            }
        }
    }
}

// ---------------------------------------------------------------- launch
extern "C" void kernel_launch(void* const* d_in, const int* in_sizes, int n_in,
                              void* d_out, int out_size, void* d_ws, size_t ws_size,
                              hipStream_t stream) {
    const float* x    = (const float*)d_in[0];
    const float* rl   = (const float*)d_in[1];
    const float* wg   = (const float*)d_in[2];
    const float* lin  = (const float*)d_in[3];
    const float* pes  = (const float*)d_in[4];
    const float* rsc  = (const float*)d_in[5];
    float*       out  = (float*)d_out;

    // workspace layout
    int*    hdr       = (int*)d_ws;          // 64 ints header
    int*    counts    = hdr;                 // [8]
    int*    base      = hdr + 8;             // [9] (base[8] = padded_total)
    int*    cursor    = hdr + 20;            // [8]
    int*    topk_idx  = hdr + 64;            // [2*NTOK]
    float*  topk_w    = (float*)(topk_idx + 2 * NTOK);
    int*    row_token = (int*)(topk_w + 2 * NTOK);
    float*  row_scale = (float*)(row_token + CAP);
    bf16_t* act       = (bf16_t*)(row_scale + CAP);   // CAP x H bf16 (16B aligned)

    hipMemsetAsync(hdr, 0, 64 * sizeof(int), stream);
    hipMemsetAsync(d_out, 0, (size_t)out_size * sizeof(float), stream);

    router_kernel<<<NTOK, 256, 0, stream>>>(x, rl, rsc, topk_idx, topk_w, counts);
    init_rows_kernel<<<(CAP + 255) / 256, 256, 0, stream>>>(row_token);
    scan_kernel<<<1, 32, 0, stream>>>(counts, base, cursor);
    assign_kernel<<<(NTOK + 255) / 256, 256, 0, stream>>>(topk_idx, topk_w, pes,
                                                          cursor, row_token, row_scale);
    dim3 g1(MT, H_DIM / 64);   // 136 x 32
    gemm_gate_kernel<<<g1, 256, 0, stream>>>(x, wg, row_token, base, act);
    dim3 g2(MT, F_DIM / 128);  // 136 x 16
    gemm_out_kernel<<<g2, 256, 0, stream>>>(act, lin, row_token, row_scale, base, out);
}
